// MultiheadAttention_76716705841261
// MI455X (gfx1250) — compile-verified, matrix operand order checked by reading, not run on hardware
//
#include <hip/hip_runtime.h>
#include <hip/hip_bf16.h>

// ---------------------------------------------------------------------------
// MI455X (gfx1250) fused structured-softmax attention.
//  * All matmuls: v_wmma_f32_16x16x32_bf16 (16384 FLOP/instr vs 2048 for the
//    f32 16x16x4 flavor; problem is ~1.1 kFLOP/byte -> compute bound).
//  * K/V tiles streamed with the Tensor Data Mover (tensor_load_to_lds,
//    TENSORcnt, ISA 08_async_tensor §7-8), double-buffered in LDS, with the
//    TDM pad feature inserting 4 dwords per 64 to de-conflict LDS banks.
//  * 304.6 KB LDS/workgroup (CDNA5 WGP has 320 KB) holds a full 16x3136 fp32
//    logit row-block so the (F=16,P=196) double softmax is single-pass.
// Workspace (d_ws): qs[3136*1024] | gate[3136*8] | mix[3136*512], fp32, ~19.4MB.
// ---------------------------------------------------------------------------

typedef __bf16 bf16_t;
typedef __attribute__((ext_vector_type(16))) __bf16 v16bf;
typedef __attribute__((ext_vector_type(8)))  float  v8f;
typedef __attribute__((ext_vector_type(4)))  unsigned int v4u;
typedef __attribute__((ext_vector_type(8)))  unsigned int v8u;

#define QN 3136
#define KN 3136
#define HN 8
#define CN 64
#define EN 512
#define FN 16
#define PN 196
#define SCALE_INV 0.125f          // 1/sqrt(C)
#define NEG_BIG  (-1.0e30f)
#define AFF_STRIDE 3144           // 3136 + pad
#define KSTRIDE 68                // 64 + 4 pad dwords inserted by TDM
#define STAGE_F (64 * KSTRIDE)    // floats per 64x64 staged tile (incl. pad)
#define SMEM_CORE ((16*AFF_STRIDE + 2*256 + 2*16*200) * 4 + 3*16*64*2)
#define SMEM_BYTES (SMEM_CORE + 4 * STAGE_F * 4)   // + K/V double buffers

// K-pair base for the 16-bit WMMA A/B per-lane layout (ISA 7.12.2):
// lanes 0-15: VGPR v<4 -> K=2v, v>=4 -> K=16+2(v-4); lanes 16-31: +8.
static __device__ __forceinline__ int kpair_base(int v, int half) {
  return ((v < 4) ? 0 : 16) + half * 8 + (v & 3) * 2;
}

static __device__ __forceinline__ v8f wmma_bf16(v16bf a, v16bf b, v8f c) {
  return __builtin_amdgcn_wmma_f32_16x16x32_bf16(false, a, false, b, (short)0,
                                                 c, false, false);
}

// ---------------------------------------------------------------------------
// Tensor Data Mover: DMA a 64-row x 64-dword tile (row stride 512 dwords in
// global) into LDS at lds_off, padding 4 dwords after every 64 so the staged
// row stride is KSTRIDE=68 floats.  D# layout per ISA 08_async_tensor §8.
// Scalar (SGPR) operands -> issue from one wave only; completion: TENSORcnt.
// ---------------------------------------------------------------------------
static __device__ __forceinline__ void tdm_load_tile64x64(unsigned lds_off,
                                                          const float* gsrc) {
  const unsigned long long ga = (unsigned long long)gsrc;
  v4u g0;
  g0[0] = 1u;                                   // count=1 (valid), user mode
  g0[1] = lds_off;                              // lds_addr[63:32 of g0]
  g0[2] = (unsigned)(ga & 0xffffffffu);         // global_addr low
  g0[3] = (unsigned)((ga >> 32) & 0x01ffffffu)  // global_addr[56:32]
        | (2u << 30);                           // type=2 ("image")
  v8u g1;
  g1[0] = (2u << 16)                            // data_size: 4 bytes
        | (1u << 20)                            // pad_enable
        | (5u << 22)                            // pad_interval: 64 dwords
        | (3u << 25);                           // pad_amount: 4 dwords
  g1[1] = 64u << 16;                            // tensor_dim0[79:48] = 64
  g1[2] = 64u << 16;                            // tensor_dim1[95:80]  = 64
  g1[3] = 64u << 16;                            // tile_dim0[127:112]  = 64
  g1[4] = 64u;                                  // tile_dim1 = 64 (tile_dim2=0)
  g1[5] = 512u;                                 // tensor_dim0_stride = 512
  g1[6] = 0u;
  g1[7] = 0u;
  const v4u gz = {0u, 0u, 0u, 0u};              // groups 2/3: dims unused
  asm volatile("tensor_load_to_lds %0, %1, %2, %3"
               :: "s"(g0), "s"(g1), "s"(gz), "s"(gz)
               : "memory");
}

// ---------------------------------------------------------------------------
// Generic WMMA GEMM with bias: out[M][Nd] = A[M][Kd] @ W[Nd][Kd]^T + bias.
// ---------------------------------------------------------------------------
__global__ void gemm_bias_wmma(const float* __restrict__ A,
                               const float* __restrict__ W,
                               const float* __restrict__ bias,
                               float* __restrict__ out,
                               int M, int Nd, int Kd) {
  const int lane = threadIdx.x & 31;
  const int wid  = threadIdx.x >> 5;
  const int ntiles = Nd >> 4;
  const int tile = blockIdx.x * 4 + wid;
  if (tile >= (M >> 4) * ntiles) return;
  const int mbase = (tile / ntiles) << 4;
  const int nbase = (tile % ntiles) << 4;
  const int half = lane >> 4;
  const int l16  = lane & 15;
  const float* arow = A + (size_t)(mbase + l16) * Kd;
  const float* wrow = W + (size_t)(nbase + l16) * Kd;

  v8f acc = {};
  for (int c0 = 0; c0 < Kd; c0 += 32) {
    v16bf av, bv;
#pragma unroll
    for (int v = 0; v < 8; ++v) {
      const int c = c0 + kpair_base(v, half);
      const float2 fa = *(const float2*)(arow + c);
      const float2 fb = *(const float2*)(wrow + c);
      av[2*v] = (bf16_t)fa.x; av[2*v+1] = (bf16_t)fa.y;
      bv[2*v] = (bf16_t)fb.x; bv[2*v+1] = (bf16_t)fb.y;
    }
    acc = wmma_bf16(av, bv, acc);
  }
  const float bn = bias[nbase + l16];
#pragma unroll
  for (int r = 0; r < 8; ++r) {   // C/D layout: lane%16 = N, vgpr r (+8*half) = M
    const int m = mbase + r + half * 8;
    out[(size_t)m * Nd + nbase + l16] = acc[r] + bn;
  }
}

// ---------------------------------------------------------------------------
// Per-key gate: g[k,h] = mask[k] ? sigmoid(-sum_c|q_coda[k,h,c]-k[k,h,c]|/8) : 0
// (reference's 2*sigmoid with the /N_ACT=2 folded in).
// ---------------------------------------------------------------------------
__global__ void gate_kernel(const float* __restrict__ qs,
                            const float* __restrict__ kmat,
                            const unsigned char* __restrict__ mask,
                            float* __restrict__ gate) {
  const int idx = blockIdx.x * blockDim.x + threadIdx.x;
  if (idx >= KN * HN) return;
  const int key = idx / HN, h = idx % HN;
  float g = 0.f;
  if (mask[key]) {
    const float* qc = qs + (size_t)key * (2 * EN) + h * (2 * CN) + CN;
    const float* kk = kmat + (size_t)key * (HN * CN) + h * CN;
    float s = 0.f;
#pragma unroll
    for (int c = 0; c < CN; ++c) s += fabsf(qc[c] - kk[c]);
    g = 1.0f / (1.0f + __expf(s * SCALE_INV));
  }
  gate[idx] = g;
}

// ---------------------------------------------------------------------------
// Fused attention core: one workgroup = (16 q-rows, 1 head), 4 wave32s.
// ---------------------------------------------------------------------------
__global__ void attn_core(const float* __restrict__ qs,
                          const float* __restrict__ kmat,
                          const float* __restrict__ vmat,
                          const unsigned char* __restrict__ mask,
                          const float* __restrict__ gate,
                          float* __restrict__ mix) {
  extern __shared__ char smem[];
  float*  aff  = (float*)smem;                 // [16][AFF_STRIDE] smax logits
  float*  fmx  = aff + 16 * AFF_STRIDE;        // [16][16]
  float*  frs  = fmx + 256;
  float*  pmx  = frs + 256;                    // [16][200]
  float*  prs  = pmx + 16 * 200;
  bf16_t* qsm  = (bf16_t*)(prs + 16 * 200);    // [16][64]
  bf16_t* qcd  = qsm + 16 * 64;                // [16][64]
  bf16_t* attS = qcd + 16 * 64;                // [16][64]
  float*  kst  = (float*)(attS + 16 * 64);     // 2 x [64][KSTRIDE] K stage
  float*  vst  = kst + 2 * STAGE_F;            // 2 x [64][KSTRIDE] V stage

  const int qt = blockIdx.x, h = blockIdx.y;
  const int qbase = qt * 16;
  const int tid  = threadIdx.x;
  const int lane = tid & 31;
  const int wave = tid >> 5;
  const int half = lane >> 4;
  const int l16  = lane & 15;

  const float* kbase_p = kmat + h * CN;        // k[key][h][c] = kbase_p[key*512+c]
  const float* vbase_p = vmat + h * CN;

  // Kick off the first K tile DMA immediately (TDM, wave 0 only).
  if (wave == 0)
    tdm_load_tile64x64((unsigned)(unsigned long long)kst, kbase_p);

  // Phase 0: stage pre-scaled q tiles into LDS as bf16.
  for (int i = tid; i < 16 * 64; i += 128) {
    const int r = i >> 6, c = i & 63;
    const float* row = qs + (size_t)(qbase + r) * (2 * EN) + h * (2 * CN);
    qsm[i] = (bf16_t)(row[c] * SCALE_INV);
    qcd[i] = (bf16_t)(row[CN + c] * SCALE_INV);
  }
  __syncthreads();

  // A operands (q rows) held in registers for the whole K sweep.
  v16bf As0, As1, Ac0, Ac1;
#pragma unroll
  for (int v = 0; v < 8; ++v) {
    const int c = kpair_base(v, half);
    As0[2*v] = qsm[l16*64 + c];      As0[2*v+1] = qsm[l16*64 + c + 1];
    As1[2*v] = qsm[l16*64 + c + 32]; As1[2*v+1] = qsm[l16*64 + c + 33];
    Ac0[2*v] = qcd[l16*64 + c];      Ac0[2*v+1] = qcd[l16*64 + c + 1];
    Ac1[2*v] = qcd[l16*64 + c + 32]; Ac1[2*v+1] = qcd[l16*64 + c + 33];
  }

  // Phase 1: stream K (TDM double-buffered), WMMA QK^T -> masked logits in LDS.
  for (int it = 0; it < KN / 64; ++it) {
    const int k0 = it * 64;
    const float* kbuf = kst + (it & 1) * STAGE_F;
    __builtin_amdgcn_s_wait_tensorcnt(0);      // current tile landed
    __syncthreads();
    if (wave == 0 && it + 1 < KN / 64)         // prefetch next tile (other buf)
      tdm_load_tile64x64((unsigned)(unsigned long long)(kst + ((it+1)&1)*STAGE_F),
                         kbase_p + (size_t)(k0 + 64) * (HN * CN));

    const int key = k0 + wave * 16 + l16;
    const float* krow = kbuf + (wave * 16 + l16) * KSTRIDE;
    v16bf B0, B1;
#pragma unroll
    for (int v = 0; v < 8; ++v) {
      const int c = kpair_base(v, half);
      const float2 f0 = *(const float2*)(krow + c);
      const float2 f1 = *(const float2*)(krow + c + 32);
      B0[2*v] = (bf16_t)f0.x; B0[2*v+1] = (bf16_t)f0.y;
      B1[2*v] = (bf16_t)f1.x; B1[2*v+1] = (bf16_t)f1.y;
    }
    v8f acc = {};
    acc = wmma_bf16(As0, B0, acc);
    acc = wmma_bf16(As1, B1, acc);
    const bool keep = mask[key] != 0;
#pragma unroll
    for (int r = 0; r < 8; ++r) {
      const int m = r + half * 8;
      aff[m * AFF_STRIDE + key] = keep ? acc[r] : NEG_BIG;
    }
    __syncthreads();                           // buf free before overwrite
  }

  // Phase 2: double softmax statistics (max + 1/sumexp), both groupings.
  for (int t = tid; t < 16 * FN; t += 128) {   // over P within f-group
    const int m = t >> 4, f = t & 15;
    const float* row = aff + m * AFF_STRIDE + f * PN;
    float mx = NEG_BIG;
    for (int j = 0; j < PN; ++j) mx = fmaxf(mx, row[j]);
    float s = 0.f;
    for (int j = 0; j < PN; ++j) s += __expf(row[j] - mx);
    fmx[t] = mx; frs[t] = 1.0f / s;
  }
  for (int t = tid; t < 16 * PN; t += 128) {   // over F at fixed p
    const int m = t / PN, p = t % PN;
    const float* col = aff + m * AFF_STRIDE + p;
    float mx = NEG_BIG;
#pragma unroll
    for (int f = 0; f < FN; ++f) mx = fmaxf(mx, col[f * PN]);
    float s = 0.f;
#pragma unroll
    for (int f = 0; f < FN; ++f) s += __expf(col[f * PN] - mx);
    pmx[m * 200 + p] = mx; prs[m * 200 + p] = 1.0f / s;
  }
  __syncthreads();

  // Phase 3: recompute coda QK^T (TDM-staged K), fuse softmax+tanh+gate into
  // bf16 att tiles, then WMMA att @ V (TDM-staged V). Wave w owns channels
  // [16w,16w+16).
  if (wave == 0) {
    tdm_load_tile64x64((unsigned)(unsigned long long)kst, kbase_p);
    tdm_load_tile64x64((unsigned)(unsigned long long)vst, vbase_p);
  }
  v8f macc = {};
  const int cb = wave * 16;
  for (int it = 0; it < KN / 64; ++it) {
    const int k0 = it * 64;
    const int buf = it & 1;
    __builtin_amdgcn_s_wait_tensorcnt(0);
    __syncthreads();
    if (wave == 0 && it + 1 < KN / 64) {
      const size_t gofs = (size_t)(k0 + 64) * (HN * CN);
      tdm_load_tile64x64((unsigned)(unsigned long long)(kst + (buf^1)*STAGE_F),
                         kbase_p + gofs);
      tdm_load_tile64x64((unsigned)(unsigned long long)(vst + (buf^1)*STAGE_F),
                         vbase_p + gofs);
    }

    const int key = k0 + wave * 16 + l16;
    const float* krow = kst + buf * STAGE_F + (wave * 16 + l16) * KSTRIDE;
    v16bf B0, B1;
#pragma unroll
    for (int v = 0; v < 8; ++v) {
      const int c = kpair_base(v, half);
      const float2 f0 = *(const float2*)(krow + c);
      const float2 f1 = *(const float2*)(krow + c + 32);
      B0[2*v] = (bf16_t)f0.x; B0[2*v+1] = (bf16_t)f0.y;
      B1[2*v] = (bf16_t)f1.x; B1[2*v+1] = (bf16_t)f1.y;
    }
    v8f cacc = {};
    cacc = wmma_bf16(Ac0, B0, cacc);
    cacc = wmma_bf16(Ac1, B1, cacc);

    const int f = key / PN, p = key % PN;
    const float gk = gate[key * HN + h];
#pragma unroll
    for (int r = 0; r < 8; ++r) {
      const int m = r + half * 8;
      const float x = aff[m * AFF_STRIDE + key];
      const float sm = __expf(x - fmx[m * 16 + f]) * frs[m * 16 + f]
                     + __expf(x - pmx[m * 200 + p]) * prs[m * 200 + p];
      const float att = 0.5f * sm + tanhf(cacc[r]) * gk;
      attS[m * 64 + wave * 16 + l16] = (bf16_t)att;
    }
    __syncthreads();                           // attS complete

    const float* vbuf = vst + buf * STAGE_F;
#pragma unroll
    for (int chunk = 0; chunk < 2; ++chunk) {  // 2 x 32-deep key chunks
      v16bf Aa, Bv;
#pragma unroll
      for (int v = 0; v < 8; ++v) {
        const int kk = chunk * 32 + kpair_base(v, half);
        Aa[2*v]   = attS[l16 * 64 + kk];
        Aa[2*v+1] = attS[l16 * 64 + kk + 1];
        Bv[2*v]   = (bf16_t)vbuf[kk * KSTRIDE + cb + l16];
        Bv[2*v+1] = (bf16_t)vbuf[(kk + 1) * KSTRIDE + cb + l16];
      }
      macc = wmma_bf16(Aa, Bv, macc);
    }
    __syncthreads();                           // attS + stage bufs free
  }

#pragma unroll
  for (int r = 0; r < 8; ++r) {
    const int m = r + half * 8;
    mix[(size_t)(qbase + m) * EN + h * CN + cb + l16] = macc[r];
  }
}

// ---------------------------------------------------------------------------
extern "C" void kernel_launch(void* const* d_in, const int* in_sizes, int n_in,
                              void* d_out, int out_size, void* d_ws, size_t ws_size,
                              hipStream_t stream) {
  const float* q   = (const float*)d_in[0];
  const float* k   = (const float*)d_in[1];
  const float* v   = (const float*)d_in[2];
  const unsigned char* m = (const unsigned char*)d_in[3];  // jnp bool -> byte
  const float* ipw = (const float*)d_in[4];
  const float* ipb = (const float*)d_in[5];
  const float* opw = (const float*)d_in[6];
  const float* opb = (const float*)d_in[7];
  float* outp = (float*)d_out;

  float* qs_ws   = (float*)d_ws;                       // [3136][1024]
  float* gate_ws = qs_ws + (size_t)QN * (2 * EN);      // [3136][8]
  float* mix_ws  = gate_ws + (size_t)KN * HN;          // [3136][512]

  // 1) in-projection: qs = q @ in_proj_w^T + b
  {
    const int tiles = (QN / 16) * ((2 * EN) / 16);
    gemm_bias_wmma<<<(tiles + 3) / 4, 128, 0, stream>>>(q, ipw, ipb, qs_ws,
                                                        QN, 2 * EN, EN);
  }
  // 2) per-key gate
  gate_kernel<<<(KN * HN + 255) / 256, 256, 0, stream>>>(qs_ws, k, m, gate_ws);
  // 3) fused attention core (304.6 KB dynamic LDS; CDNA5 WGP allows 320 KB)
  (void)hipFuncSetAttribute((const void*)attn_core,
                            hipFuncAttributeMaxDynamicSharedMemorySize,
                            SMEM_BYTES);
  attn_core<<<dim3(QN / 16, HN), 128, SMEM_BYTES, stream>>>(qs_ws, k, v, m,
                                                            gate_ws, mix_ws);
  // 4) out-projection: out = mix @ out_proj_w^T + b
  {
    const int tiles = (QN / 16) * (EN / 16);
    gemm_bias_wmma<<<(tiles + 3) / 4, 128, 0, stream>>>(mix_ws, opw, opb, outp,
                                                        QN, EN, EN);
  }
}
